// GraphAttentionLayer_44315472560876
// MI455X (gfx1250) — matrix operand order repaired and checked by reading, not compiled, not run on hardware
//
#include <hip/hip_runtime.h>
#include <hip/hip_bf16.h>

#define N_NODES 4096
#define IN_F    256
#define OUT_F   64
#define HEADS   4
#define HF      256   // HEADS*OUT_F

typedef _Float16 v16h __attribute__((ext_vector_type(16)));
typedef _Float16 v8h  __attribute__((ext_vector_type(8)));
typedef float    v8f  __attribute__((ext_vector_type(8)));
typedef int      v4i_vs __attribute__((vector_size(16)));   // matches async-LDS builtin pointee

#if __has_builtin(__builtin_amdgcn_global_load_async_to_lds_b128) && \
    __has_builtin(__builtin_amdgcn_s_wait_asynccnt)
#define GAT_HAS_ASYNC_LDS 1
#else
#define GAT_HAS_ASYNC_LDS 0
#endif

__device__ inline v16h pack16(float4 a0, float4 a1, float4 a2, float4 a3) {
    v16h v;
    v[0]=(_Float16)a0.x;  v[1]=(_Float16)a0.y;  v[2]=(_Float16)a0.z;  v[3]=(_Float16)a0.w;
    v[4]=(_Float16)a1.x;  v[5]=(_Float16)a1.y;  v[6]=(_Float16)a1.z;  v[7]=(_Float16)a1.w;
    v[8]=(_Float16)a2.x;  v[9]=(_Float16)a2.y;  v[10]=(_Float16)a2.z; v[11]=(_Float16)a2.w;
    v[12]=(_Float16)a3.x; v[13]=(_Float16)a3.y; v[14]=(_Float16)a3.z; v[15]=(_Float16)a3.w;
    return v;
}

__device__ inline v16h packB(v8h b0, v8h b1) {
    v16h v;
#pragma unroll
    for (int e = 0; e < 8; ++e) { v[e] = b0[e]; v[8 + e] = b1[e]; }
    return v;
}

// ---------------- Kernel 0: Wm (IN x HF, f32, row-major) -> WmT (HF x IN, f16) ---
__global__ __launch_bounds__(256) void gat_prep(const float* __restrict__ Wm,
                                                _Float16* __restrict__ WmT) {
    const int n = blockIdx.x;    // output column 0..255
    const int k = threadIdx.x;   // input feature 0..255
    WmT[n * IN_F + k] = (_Float16)Wm[k * HF + n];
}

// ---------------- Kernel 1: ht = h @ Wm via WMMA; write htT[nC][j] f16 ----------
// One wave computes one 16x16 tile of ht. 4096 tiles = 512 blocks * 8 waves.
__global__ __launch_bounds__(256) void gat_project(const float* __restrict__ h,
                                                   const _Float16* __restrict__ WmT,
                                                   _Float16* __restrict__ htT) {
    const int lane = threadIdx.x & 31;
    const int wave = threadIdx.x >> 5;
    const int tile = blockIdx.x * 8 + wave;    // 0..4095
    const int nt = tile & 15, mt = tile >> 4;
    const int m0 = mt * 16, n0 = nt * 16;
    const int li = lane & 15, hi = lane >> 4;

    const int row = m0 + li;   // A row loaded by this lane
    const int nC  = n0 + li;   // B column (= global feature col) for this lane

    v8f c = {};
#pragma unroll
    for (int k0 = 0; k0 < IN_F; k0 += 32) {
        // A fragment: lanes 0-15 hold K=k0..k0+7 & k0+16..k0+23; lanes 16-31 shifted by 8.
        const float* ap = h + row * IN_F + k0 + hi * 8;
        float4 a0 = ((const float4*)ap)[0];
        float4 a1 = ((const float4*)ap)[1];
        float4 a2 = ((const float4*)(ap + 16))[0];
        float4 a3 = ((const float4*)(ap + 16))[1];
        v16h A = pack16(a0, a1, a2, a3);
        // B fragment: lane li = col n; K = k0+hi*8..+7 & k0+16+hi*8..+7 (matches A).
        const _Float16* bp = WmT + nC * IN_F + k0 + hi * 8;
        v16h B = packB(((const v8h*)bp)[0], ((const v8h*)(bp + 16))[0]);
        c = __builtin_amdgcn_wmma_f32_16x16x32_f16(false, A, false, B, (short)0, c,
                                                   false, false);
    }
    // C layout: element r of lane -> (M = r + 8*hi, N = li). Write column nC of ht
    // into feature-major htT: 8 consecutive j's -> one 16B store.
    v8h hv;
#pragma unroll
    for (int r = 0; r < 8; ++r) hv[r] = (_Float16)c[r];
    *(v8h*)(htT + (size_t)nC * N_NODES + m0 + hi * 8) = hv;
}

// ---------------- Kernel 2: el[h][i] = ht_i,h . al ; er likewise (deterministic) --
__global__ __launch_bounds__(256) void gat_scores(const _Float16* __restrict__ htT,
                                                  const float* __restrict__ al,
                                                  const float* __restrict__ ar,
                                                  float* __restrict__ el,
                                                  float* __restrict__ er) {
    const int t = blockIdx.x * 256 + threadIdx.x;   // t = h*4096 + i
    const int hd = t >> 12;
    const int i  = t & (N_NODES - 1);
    float sl = 0.f, sr = 0.f;
#pragma unroll 8
    for (int f = 0; f < OUT_F; ++f) {
        float v = (float)htT[(size_t)(hd * OUT_F + f) * N_NODES + i]; // coalesced over i
        sl += v * al[f];
        sr += v * ar[f];
    }
    el[t] = sl;
    er[t] = sr;
}

// ---------------- Kernel 3: fused mask + softmax + aggregation -------------------
// Block = 16 rows x 1 head, 8 waves split the j range; WMMA: P(16x32 f16) x htT(32x64).
__global__ __launch_bounds__(256) void gat_attend(const float* __restrict__ adj,
                                                  const _Float16* __restrict__ htT,
                                                  const float* __restrict__ el,
                                                  const float* __restrict__ er,
                                                  float* __restrict__ out) {
    __shared__ float ler[N_NODES];      // this head's er vector (16 KB)
    __shared__ float lacc[8][16][64];   // per-wave partial output tiles (32 KB)
    __shared__ float ldn[8][16];        // per-wave partial denominators

    const int lane = threadIdx.x & 31;
    const int wave = threadIdx.x >> 5;
    const int li = lane & 15, hi = lane >> 4;
    const int i0 = blockIdx.x * 16;
    const int hd = blockIdx.y;

    const float* __restrict__ erh = er + hd * N_NODES;

    // Stage er[hd][0..4095] into LDS: async global->LDS (ASYNCcnt tracked),
    // else a plain cooperative copy.
#if GAT_HAS_ASYNC_LDS
    for (int t = threadIdx.x; t < N_NODES / 4; t += 256) {
        __builtin_amdgcn_global_load_async_to_lds_b128(
            (__attribute__((address_space(1))) v4i_vs*)(erh + 4 * t),
            (__attribute__((address_space(3))) v4i_vs*)(ler + 4 * t),
            0, 0);
    }
    __builtin_amdgcn_s_wait_asynccnt(0);
#else
    for (int t = threadIdx.x; t < N_NODES / 4; t += 256)
        ((float4*)ler)[t] = ((const float4*)erh)[t];
#endif
    __syncthreads();

    const float el_i = el[hd * N_NODES + i0 + li];
    const float* __restrict__ adjrow = adj + (size_t)(i0 + li) * N_NODES;
    const _Float16* __restrict__ htb = htT + (size_t)hd * OUT_F * N_NODES;

    v8f acc[4] = {v8f{}, v8f{}, v8f{}, v8f{}};
    float dsum = 0.f;

    for (int jc = wave; jc < N_NODES / 32; jc += 8) {
        const int j0 = jc * 32;
        const int jb = j0 + hi * 8;       // this lane's 8+8 K slots
        // prefetch this wave's next adjacency chunk (jc+8 -> +256 columns)
        __builtin_prefetch(adjrow + jb + 256, 0, 1);

        float4 a0 = ((const float4*)(adjrow + jb))[0];
        float4 a1 = ((const float4*)(adjrow + jb))[1];
        float4 a2 = ((const float4*)(adjrow + jb + 16))[0];
        float4 a3 = ((const float4*)(adjrow + jb + 16))[1];
        float4 e0 = ((const float4*)(ler + jb))[0];
        float4 e1 = ((const float4*)(ler + jb))[1];
        float4 e2 = ((const float4*)(ler + jb + 16))[0];
        float4 e3 = ((const float4*)(ler + jb + 16))[1];

        float pa[16] = {a0.x,a0.y,a0.z,a0.w, a1.x,a1.y,a1.z,a1.w,
                        a2.x,a2.y,a2.z,a2.w, a3.x,a3.y,a3.z,a3.w};
        float pe[16] = {e0.x,e0.y,e0.z,e0.w, e1.x,e1.y,e1.z,e1.w,
                        e2.x,e2.y,e2.z,e2.w, e3.x,e3.y,e3.z,e3.w};

        v16h A;
#pragma unroll
        for (int t = 0; t < 16; ++t) {
            float s = el_i + pe[t];
            s = s > 0.f ? s : 0.2f * s;          // leaky_relu(0.2)
            float p = pa[t] * __expf(s);         // bounded: no max-subtraction needed
            dsum += p;                           // f32 denominator
            A[t] = (_Float16)p;
        }

        const int kb = j0 + hi * 8;              // B fragment K slots (8+8) for this lane
#pragma unroll
        for (int ft = 0; ft < 4; ++ft) {
            const _Float16* bp = htb + (size_t)(ft * 16 + li) * N_NODES + kb;
            v16h B = packB(((const v8h*)bp)[0], ((const v8h*)(bp + 16))[0]);
            acc[ft] = __builtin_amdgcn_wmma_f32_16x16x32_f16(false, A, false, B,
                                                             (short)0, acc[ft],
                                                             false, false);
        }
    }

    // row denominator: lanes L and L+16 hold the two K-halves of row li
    dsum += __shfl_xor(dsum, 16, 32);
    if (lane < 16) ldn[wave][lane] = dsum;
#pragma unroll
    for (int ft = 0; ft < 4; ++ft)
#pragma unroll
        for (int r = 0; r < 8; ++r)
            lacc[wave][r + hi * 8][ft * 16 + li] = acc[ft][r];
    __syncthreads();

    // cross-wave reduce + normalize + store
    for (int t = threadIdx.x; t < 16 * OUT_F; t += 256) {
        const int i = t >> 6, f = t & (OUT_F - 1);
        float s = 0.f, d = 0.f;
#pragma unroll
        for (int w = 0; w < 8; ++w) { s += lacc[w][i][f]; d += ldn[w][i]; }
        out[(size_t)(i0 + i) * HF + hd * OUT_F + f] = s / d;
    }
}

// ---------------- Host launcher --------------------------------------------------
extern "C" void kernel_launch(void* const* d_in, const int* in_sizes, int n_in,
                              void* d_out, int out_size, void* d_ws, size_t ws_size,
                              hipStream_t stream) {
    const float* h   = (const float*)d_in[0];
    const float* adj = (const float*)d_in[1];
    const float* Wm  = (const float*)d_in[2];
    const float* al  = (const float*)d_in[3];
    const float* ar  = (const float*)d_in[4];
    float* out = (float*)d_out;

    // Workspace layout (needs ~2.26 MB):
    //   WmT f16 : 256*256*2       = 128 KB
    //   htT f16 : 256*4096*2      = 2 MB
    //   el  f32 : 4*4096*4        = 64 KB
    //   er  f32 : 4*4096*4        = 64 KB
    char* ws = (char*)d_ws;
    _Float16* WmT = (_Float16*)ws;
    _Float16* htT = (_Float16*)(ws + (131072));
    float*    el  = (float*)(ws + (131072 + 2097152));
    float*    er  = (float*)(ws + (131072 + 2097152 + 65536));

    gat_prep<<<256, 256, 0, stream>>>(Wm, WmT);
    gat_project<<<512, 256, 0, stream>>>(h, WmT, htT);
    gat_scores<<<(HEADS * N_NODES) / 256, 256, 0, stream>>>(htT, al, ar, el, er);
    gat_attend<<<dim3(N_NODES / 16, HEADS), 256, 0, stream>>>(adj, htT, el, er, out);
}
